// AttentionNet_38233798869311
// MI455X (gfx1250) — matrix operand order repaired and checked by reading, not compiled, not run on hardware
//
#include <hip/hip_runtime.h>

// ---------------------------------------------------------------------------
// Types for CDNA5 (gfx1250) wave32 WMMA: V_WMMA_F32_16X16X32_F16
// ---------------------------------------------------------------------------
typedef _Float16 half_t;
typedef __attribute__((ext_vector_type(16))) _Float16 v16h;
typedef __attribute__((ext_vector_type(8)))  float    v8f;
typedef __attribute__((ext_vector_type(4)))  _Float16 h4;

union AF { v16h v; float4 q[2]; };

__device__ __forceinline__ float4 f4zero() { return make_float4(0.f, 0.f, 0.f, 0.f); }

__device__ __forceinline__ float wave_sum32(float v) {
  for (int o = 16; o; o >>= 1) v += __shfl_xor(v, o, 32);
  return v;
}
__device__ __forceinline__ float wave_max32(float v) {
  for (int o = 16; o; o >>= 1) v = fmaxf(v, __shfl_xor(v, o, 32));
  return v;
}

// ---------------------------------------------------------------------------
// Weight conversion kernels: f32 -> f16, stored transposed (N rows x K cols)
// ---------------------------------------------------------------------------

// wq/wk/wv: (8 heads, 128 d, 16 k) -> WT[n*128 + d], n = h*16+k
__global__ void cvt_qkv_wT_kernel(const float* __restrict__ w, half_t* __restrict__ o) {
  int idx = blockIdx.x * 256 + threadIdx.x;   // 16384 total
  int n = idx >> 7;
  int d = idx & 127;
  int h = n >> 4, k = n & 15;
  o[idx] = (half_t)w[h * 2048 + d * 16 + k];
}

// wout: (8 heads, 16 v, 128 e) -> WT[e*128 + (h*16+v)]
__global__ void cvt_wout_T_kernel(const float* __restrict__ w, half_t* __restrict__ o) {
  int idx = blockIdx.x * 256 + threadIdx.x;   // 16384 total
  int c = idx & 127;
  int e = idx >> 7;
  int h = c >> 4, v = c & 15;
  o[idx] = (half_t)w[h * 2048 + v * 128 + e];
}

// generic (K,N) f32 -> (N,K) f16
__global__ void cvt_T_kernel(const float* __restrict__ w, half_t* __restrict__ o, int K, int N) {
  int idx = blockIdx.x * 256 + threadIdx.x;
  if (idx >= K * N) return;
  int n = idx / K, k = idx % K;
  o[idx] = (half_t)w[k * N + n];
}

// ---------------------------------------------------------------------------
// LayerNorm over last dim 128, f32 in -> f16 out. One wave per row.
// ---------------------------------------------------------------------------
__global__ __launch_bounds__(128)
void ln16_kernel(const float* __restrict__ x, const float* __restrict__ gw,
                 const float* __restrict__ bw, half_t* __restrict__ out, int rows) {
  int wave = threadIdx.x >> 5, lane = threadIdx.x & 31;
  int row = blockIdx.x * 4 + wave;
  if (row >= rows) return;
  const float4 xv = *(const float4*)(x + (size_t)row * 128 + lane * 4);
  float mu = wave_sum32(xv.x + xv.y + xv.z + xv.w) * (1.0f / 128.0f);
  float d0 = xv.x - mu, d1 = xv.y - mu, d2 = xv.z - mu, d3 = xv.w - mu;
  float var = wave_sum32(d0 * d0 + d1 * d1 + d2 * d2 + d3 * d3) * (1.0f / 128.0f);
  float inv = rsqrtf(var + 1e-5f);
  const float4 g4 = *(const float4*)(gw + lane * 4);
  const float4 b4 = *(const float4*)(bw + lane * 4);
  h4 o4;
  o4[0] = (half_t)(g4.x * d0 * inv + b4.x);
  o4[1] = (half_t)(g4.y * d1 * inv + b4.y);
  o4[2] = (half_t)(g4.z * d2 * inv + b4.z);
  o4[3] = (half_t)(g4.w * d3 * inv + b4.w);
  *(h4*)(out + (size_t)row * 128 + lane * 4) = o4;
}

// ---------------------------------------------------------------------------
// Generic WMMA GEMM:  out[row,col] = scale * sum_k A[row,k] * BT[col,k]
// Epilogue fully templated (no runtime flag tests in the hot unrolled code).
// One wave computes a 32x32 macro-tile (2x2 WMMA tiles, 4 independent
// accumulators to hide WMMA->WMMA hazards); 4 waves/block along N.
// ---------------------------------------------------------------------------
template <int KSTEPS, bool BIAS, bool RESID, bool O32, bool O16, bool RELU, bool OTRANS>
__global__ __launch_bounds__(128)
void gemm16_kernel(const half_t* __restrict__ A, int ldA, int sA,
                   const half_t* __restrict__ BT, int ldB, int sB,
                   const float* __restrict__ bias,
                   const float* __restrict__ resid, int ldR, int sR,
                   float* __restrict__ out32, half_t* __restrict__ out16,
                   int ldO, int sO, int M, int N, float scale) {
  const int lane = threadIdx.x & 31;
  const int wave = threadIdx.x >> 5;
  const int tn = blockIdx.x * 4 + wave;          // macro-tile along N (32 cols)
  if (tn * 32 >= N) return;                      // wave-uniform
  const int tm = blockIdx.y;                     // macro-tile along M (32 rows)
  const int z  = blockIdx.z;

  const int n = lane & 15, g = lane >> 4;
  const int koff = g ? 8 : 0;                    // A: K split {0..7,16..23}/{8..15,24..31}
  const int kb0  = g * 8;                        // B: K split {0..7,16..23}/{8..15,24..31}
  const half_t* Ap0 = A + (size_t)z * sA + (size_t)(tm * 32 + n) * ldA;
  const half_t* Ap1 = Ap0 + (size_t)16 * ldA;
  const half_t* Bp0 = BT + (size_t)z * sB + (size_t)(tn * 32 + n) * ldB;
  const half_t* Bp1 = Bp0 + (size_t)16 * ldB;

  v8f c00 = {}, c01 = {}, c10 = {}, c11 = {};
#pragma unroll
  for (int s = 0; s < KSTEPS; ++s) {
    const int kk = s * 32;
    AF a0, a1, b0, b1;
    a0.q[0] = *(const float4*)(Ap0 + kk + koff);
    a0.q[1] = *(const float4*)(Ap0 + kk + koff + 16);
    a1.q[0] = *(const float4*)(Ap1 + kk + koff);
    a1.q[1] = *(const float4*)(Ap1 + kk + koff + 16);
    b0.q[0] = *(const float4*)(Bp0 + kk + kb0);
    b0.q[1] = *(const float4*)(Bp0 + kk + kb0 + 16);
    b1.q[0] = *(const float4*)(Bp1 + kk + kb0);
    b1.q[1] = *(const float4*)(Bp1 + kk + kb0 + 16);
    c00 = __builtin_amdgcn_wmma_f32_16x16x32_f16(false, a0.v, false, b0.v, (short)0, c00, false, false);
    c01 = __builtin_amdgcn_wmma_f32_16x16x32_f16(false, a0.v, false, b1.v, (short)0, c01, false, false);
    c10 = __builtin_amdgcn_wmma_f32_16x16x32_f16(false, a1.v, false, b0.v, (short)0, c10, false, false);
    c11 = __builtin_amdgcn_wmma_f32_16x16x32_f16(false, a1.v, false, b1.v, (short)0, c11, false, false);
  }
#pragma unroll
  for (int t = 0; t < 4; ++t) {
    const v8f& c = (t == 0) ? c00 : (t == 1) ? c01 : (t == 2) ? c10 : c11;
    const int ti = t >> 1, tj = t & 1;
    const int col = tn * 32 + tj * 16 + n;
    const int row0 = tm * 32 + ti * 16 + g * 8;
    float bv = 0.f;
    if (BIAS) bv = bias[col];
#pragma unroll
    for (int r = 0; r < 8; ++r) {                // C layout: M = 8*g + r, N = lane&15
      const int row = row0 + r;
      float v = c[r] * scale + bv;
      if (RESID) v += resid[(size_t)z * sR + (size_t)row * ldR + col];
      if (RELU)  v = fmaxf(v, 0.f);
      if (O32) out32[(size_t)z * sO + (size_t)row * ldO + col] = v;
      if (O16) {
        if (OTRANS) out16[(size_t)z * sO + (size_t)col * ldO + row] = (half_t)v;
        else        out16[(size_t)z * sO + (size_t)row * ldO + col] = (half_t)v;
      }
    }
  }
}

// ---------------------------------------------------------------------------
// Fused attention: per (b, head, 16-query tile), single wave.
//   scores = scale * Q K^T (K-inner 16, A upper half zero => B upper half
//   is don't-care: unconditional loads, no divergence)        -> LDS (f32)
//   MASKED: mask words loaded up-front per tile (one clause), cndmask select
//   row softmax in-wave                                       -> LDS (f16)
//   H = A @ V via WMMA over T=512 (VT stored transposed)      -> Hout f16
// Both WMMA loops software-pipelined with modulo-rotated prefetch.
// LDS: 16*512*4 + 16*512*2 = 48KB (<< 320KB/WGP).
// ---------------------------------------------------------------------------
template <bool MASKED>
__global__ __launch_bounds__(32)
void attn_kernel(const half_t* __restrict__ Q16, const half_t* __restrict__ K16,
                 const half_t* __restrict__ VT16, const int* __restrict__ mask,
                 half_t* __restrict__ Hout, int qlen, float scale) {
  __shared__ float S[16 * 512];
  __shared__ __align__(16) half_t Ah[16 * 512];
  const int tlen = 512;
  const int lane = threadIdx.x;
  const int b = blockIdx.y >> 3, h = blockIdx.y & 7;
  const int i0 = blockIdx.x * 16;
  const int n = lane & 15, g = lane >> 4;
  const int koff = g ? 8 : 0;
  const int kb0  = g * 8;

  // Q fragment: A-matrix 16x32, K-inner = 16 (upper half zero)
  AF aq;
  {
    const half_t* p = Q16 + ((size_t)(b * qlen + i0 + n)) * 128 + h * 16 + koff;
    aq.q[0] = *(const float4*)p;
    aq.q[1] = f4zero();
  }
  const half_t* Kb = K16 + ((size_t)b * tlen) * 128 + h * 16;
  const int* mrow = MASKED ? (mask + ((size_t)(b * qlen + i0)) * tlen + n) : nullptr;

  // ---- scores (prefetch next K tile while multiplying current) ----
  AF bk;
  {
    const half_t* p = Kb + (size_t)n * 128 + koff;
    bk.q[0] = *(const float4*)p;
    bk.q[1] = *(const float4*)(p + 8);
  }
#pragma unroll 4
  for (int j0 = 0; j0 < tlen; j0 += 16) {
    AF cur = bk;
    {
      const int jn = (j0 + 16) & (tlen - 1);     // last iter re-loads tile 0 (harmless)
      const half_t* p = Kb + (size_t)(jn + n) * 128 + koff;
      bk.q[0] = *(const float4*)p;
      bk.q[1] = *(const float4*)(p + 8);
    }
    int mv[8];
    if (MASKED) {
#pragma unroll
      for (int r = 0; r < 8; ++r) mv[r] = mrow[(size_t)(g * 8 + r) * tlen + j0];
    }
    v8f c = {};
    c = __builtin_amdgcn_wmma_f32_16x16x32_f16(false, aq.v, false, cur.v,
                                               (short)0, c, false, false);
#pragma unroll
    for (int r = 0; r < 8; ++r) {
      float u = c[r] * scale;
      if (MASKED && mv[r]) u = -1e30f;
      S[(g * 8 + r) * 512 + j0 + n] = u;
    }
  }
  __syncthreads();
  // ---- row softmax (512 elems, 16 per lane) ----
  for (int row = 0; row < 16; ++row) {
    float m = -3.4e38f;
#pragma unroll
    for (int j = lane; j < tlen; j += 32) m = fmaxf(m, S[row * 512 + j]);
    m = wave_max32(m);
    float s = 0.f;
#pragma unroll
    for (int j = lane; j < tlen; j += 32) {
      float e = __expf(S[row * 512 + j] - m);
      S[row * 512 + j] = e;
      s += e;
    }
    s = wave_sum32(s);
    float inv = 1.f / s;
#pragma unroll
    for (int j = lane; j < tlen; j += 32)
      Ah[row * 512 + j] = (half_t)(S[row * 512 + j] * inv);
  }
  __syncthreads();
  // ---- H = A @ V (prefetch next A/V tiles while multiplying current) ----
  v8f acc = {};
  const half_t* Ab = &Ah[n * 512];
  const half_t* Vp = VT16 + ((size_t)(b * 128 + h * 16 + n)) * tlen;
  AF a, bv;
  a.q[0]  = *(const float4*)(Ab + koff);
  a.q[1]  = *(const float4*)(Ab + koff + 16);
  bv.q[0] = *(const float4*)(Vp + kb0);
  bv.q[1] = *(const float4*)(Vp + kb0 + 16);
#pragma unroll 4
  for (int kk = 0; kk < tlen; kk += 32) {
    AF ca = a, cb = bv;
    const int kn = (kk + 32) & (tlen - 1);
    a.q[0]  = *(const float4*)(Ab + kn + koff);
    a.q[1]  = *(const float4*)(Ab + kn + koff + 16);
    bv.q[0] = *(const float4*)(Vp + kn + kb0);
    bv.q[1] = *(const float4*)(Vp + kn + kb0 + 16);
    acc = __builtin_amdgcn_wmma_f32_16x16x32_f16(false, ca.v, false, cb.v,
                                                 (short)0, acc, false, false);
  }
#pragma unroll
  for (int r = 0; r < 8; ++r) {
    const int row = i0 + g * 8 + r;
    Hout[((size_t)(b * qlen + row)) * 128 + h * 16 + n] = (half_t)acc[r];
  }
}

// ---------------------------------------------------------------------------
// Pointer head: u = 10*tanh(U); mask -> -10000; log_softmax over T=512.
// ---------------------------------------------------------------------------
__global__ __launch_bounds__(128)
void ptr_finish_kernel(const float* __restrict__ U, const int* __restrict__ mask,
                       float* __restrict__ out, int qlen, int tlen) {
  int wave = threadIdx.x >> 5, lane = threadIdx.x & 31;
  int row = blockIdx.x * 4 + wave;
  int b = row / qlen;
  const float* Up = U + (size_t)row * tlen;
  const int* mp = mask + (size_t)b * tlen;
  float vals[16];
  float m = -3.4e38f;
#pragma unroll
  for (int j = lane, i = 0; j < tlen; j += 32, ++i) {
    float u = 10.0f * tanhf(Up[j]);
    if (mp[j]) u = -10000.0f;
    vals[i] = u;
    m = fmaxf(m, u);
  }
  m = wave_max32(m);
  float s = 0.f;
#pragma unroll
  for (int i = 0; i < 16; ++i) s += __expf(vals[i] - m);
  s = wave_sum32(s);
  float lse = m + __logf(s);
  float* op = out + (size_t)row * tlen;
#pragma unroll
  for (int j = lane, i = 0; j < tlen; j += 32, ++i) op[j] = vals[i] - lse;
}

// ---------------------------------------------------------------------------
// Host orchestration
// ---------------------------------------------------------------------------
extern "C" void kernel_launch(void* const* d_in, const int* in_sizes, int n_in,
                              void* d_out, int out_size, void* d_ws, size_t ws_size,
                              hipStream_t stream) {
  (void)in_sizes; (void)out_size;
  if (n_in < 54) return;

  const int B = 16, T = 512, QL = 128, FF = 512;
  const int MT = B * T;      // 8192
  const int MQ = B * QL;     // 2048
  const float MHA_SCALE = 0.25f;                 // 1/sqrt(16)
  const float PTR_SCALE = 0.088388347648318447f; // 1/sqrt(128)

  const float* src = (const float*)d_in[0];
  const float* tgt = (const float*)d_in[1];
  auto LP = [&](int layer, int item) -> const float* {
    return (const float*)d_in[2 + layer * 12 + item];
  };
  // items: 0 wq, 1 wk, 2 wv, 3 wout, 4 ln1_g, 5 ln1_b, 6 ln2_g, 7 ln2_b,
  //        8 w1, 9 b1, 10 w2, 11 b2.   Layers 0..2 = encoder, 3 = decoder.
  const float* ptr_wq  = (const float*)d_in[50];
  const float* ptr_wk  = (const float*)d_in[51];
  const int*   dec_msk = (const int*)d_in[52];
  const int*   ptr_msk = (const int*)d_in[53];

  // ---- workspace layout ----
  char* ws = (char*)d_ws;
  size_t off = 0;
  auto alloc = [&](size_t bytes) -> char* {
    char* p = ws + off;
    off += (bytes + 255) & ~(size_t)255;
    return p;
  };
  half_t *wqT[4], *wkT[4], *wvT[4], *woT[4], *w1T[4], *w2T[4];
  for (int l = 0; l < 4; ++l) {
    wqT[l] = (half_t*)alloc(128 * 128 * 2);
    wkT[l] = (half_t*)alloc(128 * 128 * 2);
    wvT[l] = (half_t*)alloc(128 * 128 * 2);
    woT[l] = (half_t*)alloc(128 * 128 * 2);
    w1T[l] = (half_t*)alloc(FF * 128 * 2);
    w2T[l] = (half_t*)alloc(128 * FF * 2);
  }
  half_t* pwqT = (half_t*)alloc(128 * 128 * 2);
  half_t* pwkT = (half_t*)alloc(128 * 128 * 2);
  float*  xbuf1 = (float*)alloc((size_t)MT * 128 * 4);
  float*  xbuf2 = (float*)alloc((size_t)MT * 128 * 4);
  half_t* h16   = (half_t*)alloc((size_t)MT * 128 * 2);
  half_t* Q16   = (half_t*)alloc((size_t)MT * 128 * 2);
  half_t* K16   = (half_t*)alloc((size_t)MT * 128 * 2);
  half_t* VT16  = (half_t*)alloc((size_t)MT * 128 * 2);
  half_t* Hc16  = (half_t*)alloc((size_t)MT * 128 * 2);
  float*  h2    = (float*)alloc((size_t)MT * 128 * 4);   // also reused as Up (4MB)
  half_t* ln2h  = (half_t*)alloc((size_t)MT * 128 * 2);
  half_t* ff16  = (half_t*)alloc((size_t)MT * FF * 2);
  half_t* mem16 = (half_t*)alloc((size_t)MT * 128 * 2);
  half_t* t16   = (half_t*)alloc((size_t)MQ * 128 * 2);  // also reused as dec_out16
  if (off > ws_size) return;

  // Dispatch to the exact epilogue instantiations used.
  auto gemm = [&](const half_t* A, int ldA, int sA,
                  const half_t* BT, int ldB, int sB,
                  const float* bias,
                  const float* resid, int ldR, int sR,
                  float* o32, half_t* o16, int ldO, int sO,
                  int M, int N, int K, int batch, float scale, int relu, int otrans) {
    dim3 grid((N / 32 + 3) / 4, M / 32, batch);
#define GEMM_ARGS A, ldA, sA, BT, ldB, sB, bias, resid, ldR, sR, o32, o16, ldO, sO, M, N, scale
    if (K == 128) {
      if (o16 && !o32 && !bias && !resid && !otrans)
        gemm16_kernel<4, false, false, false, true, false, false><<<grid, 128, 0, stream>>>(GEMM_ARGS);
      else if (o16 && otrans)
        gemm16_kernel<4, false, false, false, true, false, true><<<grid, 128, 0, stream>>>(GEMM_ARGS);
      else if (o32 && resid && !bias)
        gemm16_kernel<4, false, true, true, false, false, false><<<grid, 128, 0, stream>>>(GEMM_ARGS);
      else if (o16 && bias && relu)
        gemm16_kernel<4, true, false, false, true, true, false><<<grid, 128, 0, stream>>>(GEMM_ARGS);
      else  // o32 only (pointer U)
        gemm16_kernel<4, false, false, true, false, false, false><<<grid, 128, 0, stream>>>(GEMM_ARGS);
    } else {  // K == 512, always bias+resid
      if (o32 && o16)
        gemm16_kernel<16, true, true, true, true, false, false><<<grid, 128, 0, stream>>>(GEMM_ARGS);
      else if (o32)
        gemm16_kernel<16, true, true, true, false, false, false><<<grid, 128, 0, stream>>>(GEMM_ARGS);
      else
        gemm16_kernel<16, true, true, false, true, false, false><<<grid, 128, 0, stream>>>(GEMM_ARGS);
    }
#undef GEMM_ARGS
  };

  // ---- weight prep ----
  for (int l = 0; l < 4; ++l) {
    cvt_qkv_wT_kernel<<<64, 256, 0, stream>>>(LP(l, 0), wqT[l]);
    cvt_qkv_wT_kernel<<<64, 256, 0, stream>>>(LP(l, 1), wkT[l]);
    cvt_qkv_wT_kernel<<<64, 256, 0, stream>>>(LP(l, 2), wvT[l]);
    cvt_wout_T_kernel<<<64, 256, 0, stream>>>(LP(l, 3), woT[l]);
    cvt_T_kernel<<<256, 256, 0, stream>>>(LP(l, 8),  w1T[l], 128, FF);
    cvt_T_kernel<<<256, 256, 0, stream>>>(LP(l, 10), w2T[l], FF, 128);
  }
  cvt_T_kernel<<<64, 256, 0, stream>>>(ptr_wq, pwqT, 128, 128);
  cvt_T_kernel<<<64, 256, 0, stream>>>(ptr_wk, pwkT, 128, 128);

  // ---- encoder (3 layers) ----
  const float* x = src;
  for (int L = 0; L < 3; ++L) {
    float* y = (L == 1) ? xbuf2 : xbuf1;
    ln16_kernel<<<MT / 4, 128, 0, stream>>>(x, LP(L, 4), LP(L, 5), h16, MT);
    gemm(h16, 128, 0, wqT[L], 128, 0, nullptr, nullptr, 0, 0,
         nullptr, Q16, 128, 0, MT, 128, 128, 1, 1.f, 0, 0);
    gemm(h16, 128, 0, wkT[L], 128, 0, nullptr, nullptr, 0, 0,
         nullptr, K16, 128, 0, MT, 128, 128, 1, 1.f, 0, 0);
    // V projection with fused transpose: VT16[b, e, t] (batched over b)
    gemm(h16, 128, T * 128, wvT[L], 128, 0, nullptr, nullptr, 0, 0,
         nullptr, VT16, T, 128 * T, T, 128, 128, B, 1.f, 0, 1);
    attn_kernel<false><<<dim3(T / 16, B * 8), 32, 0, stream>>>(Q16, K16, VT16, nullptr,
                                                               Hc16, T, MHA_SCALE);
    gemm(Hc16, 128, 0, woT[L], 128, 0, nullptr, x, 128, 0,
         h2, nullptr, 128, 0, MT, 128, 128, 1, 1.f, 0, 0);
    ln16_kernel<<<MT / 4, 128, 0, stream>>>(h2, LP(L, 6), LP(L, 7), ln2h, MT);
    gemm(ln2h, 128, 0, w1T[L], 128, 0, LP(L, 9), nullptr, 0, 0,
         nullptr, ff16, FF, 0, MT, FF, 128, 1, 1.f, 1, 0);
    gemm(ff16, FF, 0, w2T[L], FF, 0, LP(L, 11), h2, 128, 0,
         y, (L == 2) ? mem16 : nullptr, 128, 0, MT, 128, 512, 1, 1.f, 0, 0);
    x = y;
  }
  const float* memf = x;  // xbuf1

  // ---- decoder (layer 3), shared ln1 for tgt and memory ----
  ln16_kernel<<<MQ / 4, 128, 0, stream>>>(tgt, LP(3, 4), LP(3, 5), t16, MQ);
  ln16_kernel<<<MT / 4, 128, 0, stream>>>(memf, LP(3, 4), LP(3, 5), h16, MT); // m16
  gemm(t16, 128, 0, wqT[3], 128, 0, nullptr, nullptr, 0, 0,
       nullptr, Q16, 128, 0, MQ, 128, 128, 1, 1.f, 0, 0);
  gemm(h16, 128, 0, wkT[3], 128, 0, nullptr, nullptr, 0, 0,
       nullptr, K16, 128, 0, MT, 128, 128, 1, 1.f, 0, 0);
  gemm(h16, 128, T * 128, wvT[3], 128, 0, nullptr, nullptr, 0, 0,
       nullptr, VT16, T, 128 * T, T, 128, 128, B, 1.f, 0, 1);
  attn_kernel<true><<<dim3(QL / 16, B * 8), 32, 0, stream>>>(Q16, K16, VT16, dec_msk,
                                                             Hc16, QL, MHA_SCALE);
  gemm(Hc16, 128, 0, woT[3], 128, 0, nullptr, tgt, 128, 0,
       h2, nullptr, 128, 0, MQ, 128, 128, 1, 1.f, 0, 0);
  ln16_kernel<<<MQ / 4, 128, 0, stream>>>(h2, LP(3, 6), LP(3, 7), ln2h, MQ);
  gemm(ln2h, 128, 0, w1T[3], 128, 0, LP(3, 9), nullptr, 0, 0,
       nullptr, ff16, FF, 0, MQ, FF, 128, 1, 1.f, 1, 0);
  gemm(ff16, FF, 0, w2T[3], FF, 0, LP(3, 11), h2, 128, 0,
       nullptr, t16 /* dec_out16 */, 128, 0, MQ, 128, 512, 1, 1.f, 0, 0);

  // ---- pointer head ----
  gemm(t16, 128, 0, pwqT, 128, 0, nullptr, nullptr, 0, 0,
       nullptr, Q16, 128, 0, MQ, 128, 128, 1, 1.f, 0, 0);              // Qp
  gemm(mem16, 128, 0, pwkT, 128, 0, nullptr, nullptr, 0, 0,
       nullptr, K16, 128, 0, MT, 128, 128, 1, 1.f, 0, 0);              // Kp
  // Up[b,q,t] = PTR_SCALE * Qp[b,q,:] . Kp[b,t,:]  (batched, reuse h2 as Up)
  gemm(Q16, 128, QL * 128, K16, 128, T * 128, nullptr,
       nullptr, 0, 0, h2, nullptr, T, QL * T,
       QL, T, 128, B, PTR_SCALE, 0, 0);
  ptr_finish_kernel<<<MQ / 4, 128, 0, stream>>>(h2, ptr_msk, (float*)d_out, QL, T);
}
